// GINe_85899345974
// MI455X (gfx1250) — matrix operand order repaired and checked by reading, not compiled, not run on hardware
//
#include <hip/hip_runtime.h>
#include <hip/hip_bf16.h>

typedef __attribute__((ext_vector_type(16))) _Float16 v16h;
typedef __attribute__((ext_vector_type(8)))  float    v8f;

#define HB 128
#define BN_EPS_F 1e-5f

union HalfPack { _Float16 h[2]; unsigned int u; };
union FragU    { uint4 q[2]; v16h h; };

// ---------------- utility kernels ----------------

__global__ void fzero_kernel(float* p, long n) {
  long i = (long)blockIdx.x * blockDim.x + threadIdx.x;
  if (i < n) p[i] = 0.0f;
}

// Pack a row-major f32 weight [K][Ncol] into f16 WMMA B-fragment order:
// tiles (kt, nt), within tile: lane-major (32 lanes x 16 halves).
// lane l (0..31): column n = nt*16 + (l & 15); element e: k = kt*32 + ((l<16)?0:16) + e
__global__ void pack_w_kernel(const float* __restrict__ src, _Float16* __restrict__ dst,
                              int K, int Ncol, int ntN, long total) {
  long t = (long)blockIdx.x * blockDim.x + threadIdx.x;
  if (t >= total) return;
  long tile = t >> 9;
  int  r    = (int)(t & 511);
  int  lane = r >> 4;
  int  e    = r & 15;
  int  kt   = (int)(tile / ntN);
  int  nt   = (int)(tile % ntN);
  int  n    = nt * 16 + (lane & 15);
  int  k    = kt * 32 + ((lane < 16) ? 0 : 16) + e;
  float v = (k < K && n < Ncol) ? src[(long)k * Ncol + n] : 0.0f;
  dst[t] = (_Float16)v;
}

__global__ void pack_b_kernel(const float* __restrict__ src, float* __restrict__ dst,
                              int n, int np) {
  int i = blockIdx.x * blockDim.x + threadIdx.x;
  if (i < np) dst[i] = (i < n) ? src[i] : 0.0f;
}

// ---------------- WMMA GEMM ----------------
// MODE 0: A = direct row-major [rows][lda]
// MODE 1: A row j = [ h[src[j]][0..127] | h[dst[j]][0..127] | e[j][0..127] ], K = 384
// RELU: relu on output; HALFADD: C += 0.5*(result); GRELU: relu gathered node parts (MODE 1)
template<int MODE, int NT, bool RELU, bool HALFADD, bool GRELU>
__global__ __launch_bounds__(256) void gemm_kernel(
    const float* __restrict__ A, long lda,
    const float* __restrict__ hsrc, const float* __restrict__ esrc,
    const int* __restrict__ srcIdx, const int* __restrict__ dstIdx,
    const _Float16* __restrict__ W, const float* __restrict__ bias,
    float* __restrict__ C, long ldc, long rows, int kchunks, int storeN)
{
  __shared__ _Float16 As[128 * 32];   // 128 rows x 32 k, f16 (8 KB)
  const int  tid      = threadIdx.x;
  const long blockRow = (long)blockIdx.x * 128;
  const int  lane     = tid & 31;
  const int  rbase    = (tid >> 5) * 16;   // wave's 16-row slice

  v8f zf = {};
  v8f acc[NT];
#pragma unroll
  for (int i = 0; i < NT; ++i) acc[i] = zf;

  const int  rowLocal = tid >> 1;
  const int  half     = tid & 1;
  const long jrow     = blockRow + rowLocal;

  for (int kt = 0; kt < kchunks; ++kt) {
    __syncthreads();
    // ---- cooperative A-tile load: f32 global -> f16 LDS ----
    float f[16];
    if (jrow < rows) {
      const float* sp;
      bool grelu = false;
      if (MODE == 0) {
        sp = A + jrow * lda + (long)kt * 32 + half * 16;
      } else {
        int  part = kt >> 2;
        long coff = (long)(kt & 3) * 32 + half * 16;
        if (part == 0)      { sp = hsrc + (long)srcIdx[jrow] * HB + coff; grelu = GRELU; }
        else if (part == 1) { sp = hsrc + (long)dstIdx[jrow] * HB + coff; grelu = GRELU; }
        else                { sp = esrc + jrow * HB + coff; }
      }
      const float4* s4 = (const float4*)sp;
      float4 a0 = s4[0], a1 = s4[1], a2 = s4[2], a3 = s4[3];
      f[0]=a0.x;  f[1]=a0.y;  f[2]=a0.z;  f[3]=a0.w;
      f[4]=a1.x;  f[5]=a1.y;  f[6]=a1.z;  f[7]=a1.w;
      f[8]=a2.x;  f[9]=a2.y;  f[10]=a2.z; f[11]=a2.w;
      f[12]=a3.x; f[13]=a3.y; f[14]=a3.z; f[15]=a3.w;
      if (grelu) {
#pragma unroll
        for (int i = 0; i < 16; ++i) f[i] = fmaxf(f[i], 0.0f);
      }
    } else {
#pragma unroll
      for (int i = 0; i < 16; ++i) f[i] = 0.0f;
    }
    unsigned int u[8];
#pragma unroll
    for (int i = 0; i < 8; ++i) {
      HalfPack p; p.h[0] = (_Float16)f[2*i]; p.h[1] = (_Float16)f[2*i+1]; u[i] = p.u;
    }
    uint4* dlds = (uint4*)&As[rowLocal * 32 + half * 16];
    dlds[0] = make_uint4(u[0], u[1], u[2], u[3]);
    dlds[1] = make_uint4(u[4], u[5], u[6], u[7]);
    __syncthreads();

    // ---- preload ALL B fragments for this k-step (one load clause, staged waits) ----
    FragU fb[NT];
#pragma unroll
    for (int nt = 0; nt < NT; ++nt) {
      const uint4* wp = (const uint4*)(W + (((long)kt * NT + nt) * 32 + lane) * 16);
      fb[nt].q[0] = wp[0];
      fb[nt].q[1] = wp[1];
    }

    // ---- A fragment (contiguous k order, same as packed W: lanes 0-15 K{0..15}, lanes 16-31 K{16..31}) ----
    FragU fa;
    const _Float16* ap = &As[(rbase + (lane & 15)) * 32 + ((lane >> 4) * 16)];
    fa.q[0] = *(const uint4*)ap;         // K = base .. base+7
    fa.q[1] = *(const uint4*)(ap + 8);   // K = base+8 .. base+15

    // ---- back-to-back WMMAs ----
#pragma unroll
    for (int nt = 0; nt < NT; ++nt) {
      acc[nt] = __builtin_amdgcn_wmma_f32_16x16x32_f16(
          false, fa.h, false, fb[nt].h, (short)0, acc[nt], false, false);
    }
  }

  // ---- epilogue: C layout lane<16 -> rows 0..7, lane>=16 -> rows 8..15; col = nt*16 + (lane&15)
  const int rofs = (lane >> 4) * 8;
#pragma unroll
  for (int nt = 0; nt < NT; ++nt) {
    int   col = nt * 16 + (lane & 15);
    float b   = bias[col];
#pragma unroll
    for (int r = 0; r < 8; ++r) {
      long row = blockRow + rbase + rofs + r;
      if (col < storeN && row < rows) {
        float v = acc[nt][r] + b;
        if (RELU) v = fmaxf(v, 0.0f);
        long o = row * ldc + col;
        if (HALFADD) C[o] += 0.5f * v; else C[o] = v;
      }
    }
  }
}

// ---------------- GNN glue kernels ----------------

// msg = relu(h[src] + linE); agg[dst] += msg  (atomic, L2-resident)
__global__ void scatter_msg_kernel(const float* __restrict__ hp, const float* __restrict__ tE,
                                   const int* __restrict__ srcI, const int* __restrict__ dstI,
                                   float* __restrict__ agg, long nE) {
  long idx = (long)blockIdx.x * blockDim.x + threadIdx.x;
  if (idx >= nE * HB) return;
  long j = idx >> 7;
  int  f = (int)(idx & (HB - 1));
  float v = hp[(long)srcI[j] * HB + f] + tE[idx];
  v = fmaxf(v, 0.0f);
  if (v != 0.0f) atomicAdd(&agg[(long)dstI[j] * HB + f], v);
}

__global__ void add_kernel(float* __restrict__ dst, const float* __restrict__ src, long n) {
  long i = (long)blockIdx.x * blockDim.x + threadIdx.x;
  if (i < n) dst[i] += src[i];
}

__global__ void bn_partial_kernel(const float* __restrict__ z, float* __restrict__ gsum,
                                  float* __restrict__ gsumsq, long n) {
  int  f      = threadIdx.x & 127;
  long start  = (long)blockIdx.x * 2 + (threadIdx.x >> 7);
  long stride = (long)gridDim.x * 2;
  float s = 0.0f, ss = 0.0f;
  for (long r = start; r < n; r += stride) {
    float v = z[r * HB + f];
    s += v; ss += v * v;
  }
  atomicAdd(&gsum[f], s);
  atomicAdd(&gsumsq[f], ss);
}

__global__ void bn_final_kernel(const float* gsum, const float* gsumsq,
                                float* mu, float* rsig, long n) {
  int f = threadIdx.x;
  if (f < HB) {
    float m = gsum[f] / (float)n;
    float v = gsumsq[f] / (float)n - m * m;
    mu[f]   = m;
    rsig[f] = rsqrtf(v + BN_EPS_F);
  }
}

// h = (h + relu((z - mu)*rsig*gamma + beta)) / 2
__global__ void bn_apply_kernel(float* __restrict__ hp, const float* __restrict__ z,
                                const float* __restrict__ mu, const float* __restrict__ rsig,
                                const float* __restrict__ gamma, const float* __restrict__ beta,
                                long n128) {
  long i = (long)blockIdx.x * blockDim.x + threadIdx.x;
  if (i >= n128) return;
  int f = (int)(i & (HB - 1));
  float y = (z[i] - mu[f]) * rsig[f] * gamma[f] + beta[f];
  hp[i] = (hp[i] + fmaxf(y, 0.0f)) * 0.5f;
}

// ---------------- host launch ----------------

extern "C" void kernel_launch(void* const* d_in, const int* in_sizes, int n_in,
                              void* d_out, int out_size, void* d_ws, size_t ws_size,
                              hipStream_t stream) {
  const float* x      = (const float*)d_in[0];
  const int*   eidx   = (const int*)  d_in[1];
  const float* eattr  = (const float*)d_in[2];
  const float* node_w = (const float*)d_in[3];
  const float* node_b = (const float*)d_in[4];
  const float* edge_w = (const float*)d_in[5];
  const float* edge_b = (const float*)d_in[6];
  const float* lin_w  = (const float*)d_in[7];
  const float* lin_b  = (const float*)d_in[8];
  const float* w1     = (const float*)d_in[9];
  const float* b1     = (const float*)d_in[10];
  const float* w2     = (const float*)d_in[11];
  const float* b2     = (const float*)d_in[12];
  const float* gamma_ = (const float*)d_in[13];
  const float* beta_  = (const float*)d_in[14];
  const float* ew1    = (const float*)d_in[15];
  const float* eb1    = (const float*)d_in[16];
  const float* ew2    = (const float*)d_in[17];
  const float* eb2    = (const float*)d_in[18];
  const float* mw1    = (const float*)d_in[19];
  const float* mb1    = (const float*)d_in[20];
  const float* mw2    = (const float*)d_in[21];
  const float* mb2    = (const float*)d_in[22];
  const float* mw3    = (const float*)d_in[23];
  const float* mb3    = (const float*)d_in[24];

  const long Nn = in_sizes[0] / 128;   // nodes
  const long Ee = in_sizes[2] / 64;    // edges
  const int* srcI = eidx;
  const int* dstI = eidx + Ee;

  // ---- workspace carving ----
  size_t off = 0;
  auto alloc = [&](size_t bytes) -> void* {
    void* p = (char*)d_ws + off;
    off += (bytes + 255) & ~(size_t)255;
    return p;
  };
  float* hp  = (float*)alloc((size_t)Nn * HB * 4);
  float* agg = (float*)alloc((size_t)Nn * HB * 4);
  float* zb  = (float*)alloc((size_t)Nn * HB * 4);
  float* ep  = (float*)alloc((size_t)Ee * HB * 4);
  float* tE  = (float*)alloc((size_t)Ee * HB * 4);
  float* tE2 = (float*)alloc((size_t)Ee * 32 * 4);

  _Float16* node_wP = (_Float16*)alloc(128 * 128 * 2);
  _Float16* edge_wP = (_Float16*)alloc(64 * 128 * 2);
  _Float16* lin_wP[2]; _Float16* w1P[2]; _Float16* w2P[2];
  _Float16* ew1P[2];   _Float16* ew2P[2];
  for (int i = 0; i < 2; ++i) {
    lin_wP[i] = (_Float16*)alloc(128 * 128 * 2);
    w1P[i]    = (_Float16*)alloc(128 * 128 * 2);
    w2P[i]    = (_Float16*)alloc(128 * 128 * 2);
    ew1P[i]   = (_Float16*)alloc(384 * 128 * 2);
    ew2P[i]   = (_Float16*)alloc(128 * 128 * 2);
  }
  _Float16* mw1P = (_Float16*)alloc(384 * 64 * 2);
  _Float16* mw2P = (_Float16*)alloc(64 * 32 * 2);
  _Float16* mw3P = (_Float16*)alloc(32 * 16 * 2);

  float* node_bP = (float*)alloc(128 * 4);
  float* edge_bP = (float*)alloc(128 * 4);
  float* lin_bP[2]; float* b1P[2]; float* b2P[2]; float* eb1P[2]; float* eb2P[2];
  float* gammaP[2]; float* betaP[2];
  for (int i = 0; i < 2; ++i) {
    lin_bP[i] = (float*)alloc(128 * 4);
    b1P[i]    = (float*)alloc(128 * 4);
    b2P[i]    = (float*)alloc(128 * 4);
    eb1P[i]   = (float*)alloc(128 * 4);
    eb2P[i]   = (float*)alloc(128 * 4);
    gammaP[i] = (float*)alloc(128 * 4);
    betaP[i]  = (float*)alloc(128 * 4);
  }
  float* mb1P = (float*)alloc(64 * 4);
  float* mb2P = (float*)alloc(32 * 4);
  float* mb3P = (float*)alloc(16 * 4);
  float* gsum   = (float*)alloc(128 * 4);
  float* gsumsq = (float*)alloc(128 * 4);
  float* muB    = (float*)alloc(128 * 4);
  float* rsigB  = (float*)alloc(128 * 4);

  auto packW = [&](const float* s, _Float16* d, int K, int Ncol, int Kp, int Np) {
    long total = (long)(Kp / 32) * (Np / 16) * 512;
    pack_w_kernel<<<(unsigned)((total + 255) / 256), 256, 0, stream>>>(s, d, K, Ncol, Np / 16, total);
  };
  auto packB = [&](const float* s, float* d, int n, int np) {
    pack_b_kernel<<<(np + 255) / 256, 256, 0, stream>>>(s, d, n, np);
  };

  // ---- pack all parameters (zero-padded) ----
  packW(node_w, node_wP, 128, 100, 128, 128);  packB(node_b, node_bP, 100, 128);
  packW(edge_w, edge_wP,  64, 100,  64, 128);  packB(edge_b, edge_bP, 100, 128);
  for (int i = 0; i < 2; ++i) {
    packW(lin_w + (long)i * 100 * 100, lin_wP[i], 100, 100, 128, 128);
    packW(w1    + (long)i * 100 * 100, w1P[i],    100, 100, 128, 128);
    packW(w2    + (long)i * 100 * 100, w2P[i],    100, 100, 128, 128);
    packW(ew1   + (long)i * 300 * 100, ew1P[i],   300, 100, 384, 128);
    packW(ew2   + (long)i * 100 * 100, ew2P[i],   100, 100, 128, 128);
    packB(lin_b + i * 100, lin_bP[i], 100, 128);
    packB(b1    + i * 100, b1P[i],    100, 128);
    packB(b2    + i * 100, b2P[i],    100, 128);
    packB(eb1   + i * 100, eb1P[i],   100, 128);
    packB(eb2   + i * 100, eb2P[i],   100, 128);
    packB(gamma_+ i * 100, gammaP[i], 100, 128);
    packB(beta_ + i * 100, betaP[i],  100, 128);
  }
  packW(mw1, mw1P, 300, 50, 384, 64);  packB(mb1, mb1P, 50, 64);
  packW(mw2, mw2P,  50, 25,  64, 32);  packB(mb2, mb2P, 25, 32);
  packW(mw3, mw3P,  25,  2,  32, 16);  packB(mb3, mb3P,  2, 16);

  auto gblk = [](long rows) { return (unsigned)((rows + 127) / 128); };
  const unsigned gN = gblk(Nn), gE = gblk(Ee);
  const long Nh = Nn * HB, Eh = Ee * HB;
  const unsigned zN = (unsigned)((Nh + 255) / 256);
  const unsigned aE = (unsigned)((Eh + 255) / 256);

  // ---- input embeddings ----
  gemm_kernel<0,8,false,false,false><<<gN, 256, 0, stream>>>(
      x, 128, nullptr, nullptr, nullptr, nullptr, node_wP, node_bP, hp, HB, Nn, 4, 128);
  gemm_kernel<0,8,false,false,false><<<gE, 256, 0, stream>>>(
      eattr, 64, nullptr, nullptr, nullptr, nullptr, edge_wP, edge_bP, ep, HB, Ee, 2, 128);

  // ---- GNN layers ----
  for (int i = 0; i < 2; ++i) {
    // linE = e @ lin_w + lin_b
    gemm_kernel<0,8,false,false,false><<<gE, 256, 0, stream>>>(
        ep, HB, nullptr, nullptr, nullptr, nullptr, lin_wP[i], lin_bP[i], tE, HB, Ee, 4, 128);
    // agg = segment_sum(relu(h[src] + linE), dst)
    fzero_kernel<<<zN, 256, 0, stream>>>(agg, Nh);
    scatter_msg_kernel<<<aE, 256, 0, stream>>>(hp, tE, srcI, dstI, agg, Ee);
    // z_in = h + agg
    add_kernel<<<zN, 256, 0, stream>>>(agg, hp, Nh);
    // z = relu(z_in @ w1 + b1) @ w2 + b2
    gemm_kernel<0,8,true,false,false><<<gN, 256, 0, stream>>>(
        agg, HB, nullptr, nullptr, nullptr, nullptr, w1P[i], b1P[i], zb, HB, Nn, 4, 128);
    gemm_kernel<0,8,false,false,false><<<gN, 256, 0, stream>>>(
        zb, HB, nullptr, nullptr, nullptr, nullptr, w2P[i], b2P[i], agg, HB, Nn, 4, 128);
    // BatchNorm (training-mode, biased var) + residual
    fzero_kernel<<<1, 256, 0, stream>>>(gsum, 256);  // gsum+gsumsq contiguous
    bn_partial_kernel<<<512, 256, 0, stream>>>(agg, gsum, gsumsq, Nn);
    bn_final_kernel<<<1, 128, 0, stream>>>(gsum, gsumsq, muB, rsigB, Nn);
    bn_apply_kernel<<<zN, 256, 0, stream>>>(hp, agg, muB, rsigB, gammaP[i], betaP[i], Nh);
    // edge update: e += 0.5*(relu(cat @ ew1 + eb1) @ ew2 + eb2)
    gemm_kernel<1,8,true,false,false><<<gE, 256, 0, stream>>>(
        nullptr, 0, hp, ep, srcI, dstI, ew1P[i], eb1P[i], tE, HB, Ee, 12, 128);
    gemm_kernel<0,8,false,true,false><<<gE, 256, 0, stream>>>(
        tE, HB, nullptr, nullptr, nullptr, nullptr, ew2P[i], eb2P[i], ep, HB, Ee, 4, 128);
  }

  // ---- final edge MLP: feat = [relu(h_src,h_dst), e] ----
  gemm_kernel<1,4,true,false,true><<<gE, 256, 0, stream>>>(
      nullptr, 0, hp, ep, srcI, dstI, mw1P, mb1P, tE, 64, Ee, 12, 64);
  gemm_kernel<0,2,true,false,false><<<gE, 256, 0, stream>>>(
      tE, 64, nullptr, nullptr, nullptr, nullptr, mw2P, mb2P, tE2, 32, Ee, 2, 32);
  gemm_kernel<0,1,false,false,false><<<gE, 256, 0, stream>>>(
      tE2, 32, nullptr, nullptr, nullptr, nullptr, mw3P, mb3P, (float*)d_out, 2, Ee, 1, 2);
}
